// SPRM_8108898254865
// MI455X (gfx1250) — matrix operand order repaired and checked by reading, not compiled, not run on hardware
//
#include <hip/hip_runtime.h>

typedef __attribute__((ext_vector_type(16))) __bf16 v16bf;
typedef __attribute__((ext_vector_type(8)))  __bf16 v8bf;
typedef __attribute__((ext_vector_type(4)))  __bf16 v4bf;
typedef __attribute__((ext_vector_type(8)))  float  v8f;
typedef __attribute__((ext_vector_type(4)))  float  v4f;

// ---------------------------------------------------------------------------
// WMMA helpers (CDNA5 wave32, 16x16x32 bf16 -> f32)
// A layout: lane m=lane&15 holds row m; lanes 0-15: K = 0..15 (kb=0);
//           lanes 16-31 same rows, K = 16..31 (kb=16).
// B layout: weights stored N-major (N x K row-major): lane n=lane&15 holds
//           col n; lanes 0-15 read 16 contiguous halves at K=k0, lanes 16-31
//           at K=k0+16.
// C layout: vgpr j -> row = j + (lane>>4)*8, col = lane&15.
// ---------------------------------------------------------------------------
__device__ __forceinline__ v8f wmma_bf16(v16bf a, v16bf b, v8f c) {
  return __builtin_amdgcn_wmma_f32_16x16x32_bf16(false, a, false, b, (short)0, c,
                                                 false, false);
}

__device__ __forceinline__ v16bf load_a16(const __bf16* p, int kb) {
  v8bf lo = *(const v8bf*)(p + kb);
  v8bf hi = *(const v8bf*)(p + kb + 8);
  return __builtin_shufflevector(lo, hi, 0, 1, 2, 3, 4, 5, 6, 7,
                                         8, 9, 10, 11, 12, 13, 14, 15);
}

#define ZERO8x(A, N)                         \
  _Pragma("unroll") for (int _t = 0; _t < (N); ++_t) \
  _Pragma("unroll") for (int _j = 0; _j < 8; ++_j) (A)[_t][_j] = 0.f;

// ---------------------------------------------------------------------------
// Prep kernels (tiny)
// ---------------------------------------------------------------------------
__global__ void sprm_prep_pat(const float* __restrict__ pat,
                              float* __restrict__ patln) {
  int row = blockIdx.x * blockDim.x + threadIdx.x;
  if (row >= 512) return;
  const float* p = pat + (size_t)row * 256;
  float s = 0.f, ss = 0.f;
  for (int i = 0; i < 256; ++i) { float v = p[i]; s += v; ss += v * v; }
  float mu  = s * (1.f / 256.f);
  float var = ss * (1.f / 256.f) - mu * mu;
  float rs  = rsqrtf(var + 1e-5f);
  // layernorm(layernorm(x)): second LN has mean 0, var = var*rs^2 exactly
  float rs2 = rsqrtf(var * rs * rs + 1e-5f);
  float sc  = rs * rs2;
  float* o = patln + (size_t)row * 256;
  for (int i = 0; i < 256; ++i) o[i] = (p[i] - mu) * sc;
}

__global__ void sprm_prep_kv(const float* __restrict__ patln,
                             const float* __restrict__ wkv,
                             __bf16* __restrict__ kbuf,   // [h][w=512][d=64]
                             __bf16* __restrict__ vbuf) { // [h][d=64][w=512]
  int idx = blockIdx.x * blockDim.x + threadIdx.x;
  if (idx >= 512 * 1024) return;
  int w = idx >> 10;
  int nn = idx & 1023;
  const float* a = patln + (size_t)w * 256;
  const float* b = wkv + (size_t)nn * 256;
  float s = 0.f;
  for (int i = 0; i < 256; ++i) s += a[i] * b[i];
  int h = nn >> 7, j = nn & 127;
  if (j < 64) kbuf[((size_t)h * 512 + w) * 64 + j] = (__bf16)s;
  else        vbuf[((size_t)h * 64 + (j - 64)) * 512 + w] = (__bf16)s;
}

__global__ void sprm_prep_wt(const float* __restrict__ convw,  // (256,256,3)
                             const float* __restrict__ wq,     // (512,256)
                             const float* __restrict__ wout,   // (256,512)
                             __bf16* __restrict__ wcat,        // (256,768)
                             __bf16* __restrict__ wqb,
                             __bf16* __restrict__ woutb) {
  int idx = blockIdx.x * blockDim.x + threadIdx.x;
  const int NC = 256 * 768, NQ = 512 * 256, NO = 256 * 512;
  if (idx < NC) {
    int o = idx / 768, k = idx % 768;
    int tau = k >> 8, i = k & 255;
    wcat[idx] = (__bf16)convw[((size_t)o * 256 + i) * 3 + tau];
  } else if (idx < NC + NQ) {
    int t = idx - NC;
    wqb[t] = (__bf16)wq[t];
  } else if (idx < NC + NQ + NO) {
    int t = idx - NC - NQ;
    woutb[t] = (__bf16)wout[t];
  }
}

// ---------------------------------------------------------------------------
// Fused main kernel: conv + LN + Q + attention + out-proj + LN
// 64 rows per block, 256 threads (8 waves): wave = (strip 0..3, grp 0..1)
// ---------------------------------------------------------------------------
__global__ __launch_bounds__(256)
void sprm_fused(const float* __restrict__ x,
                const float* __restrict__ convb,
                const float* __restrict__ boutp,
                const __bf16* __restrict__ wcat,
                const __bf16* __restrict__ wqb,
                const __bf16* __restrict__ kbuf,
                const __bf16* __restrict__ vbuf,
                const __bf16* __restrict__ woutb,
                float* __restrict__ outp) {
  __shared__ __align__(16) float  sR0[64 * 260];   // y(f32) -> scores(bf16) -> out(f32)
  __shared__ __align__(16) __bf16 sR1[66 * 264];   // x-window -> qh + red + inv
  __shared__ __align__(16) __bf16 sXN[64 * 264];   // normalized conv output
  __shared__ __align__(16) __bf16 sATT[64 * 520];  // attention output (all heads)

  const int tid = threadIdx.x;
  const int lane = tid & 31;
  const int wv = tid >> 5;
  const int strip = wv & 3;   // 16-row strip
  const int grp = wv >> 2;    // N-half / K-half selector
  const int m = lane & 15;
  const int h16 = lane >> 4;
  const int kbA = h16 * 16;
  const int kbB = h16 * 16;

  const int r0 = blockIdx.x * 64;
  const int p0 = r0 & (2048 - 1);

  __bf16* sXW = sR1;
  __bf16* sQH = sR1;                         // 64 x 72 halves   [0, 9216) B
  float* sRED = (float*)(sR1 + 64 * 72);     // 64 x 68 f32      [9216, 26624) B
  float* sInv = (float*)(sR1 + 13312);       // 64 f32           [26624, 26880) B
  __bf16* sS = (__bf16*)sR0;                 // 64 x 520 halves

  // ---- stage 1: x window (66 rows; 2 rows of left pad at sequence start) ----
  for (int idx = tid; idx < 66 * 64; idx += 256) {
    int wr = idx >> 6;
    int ci = (idx & 63) * 4;
    int pr = p0 - 2 + wr;
    v4f v = {0.f, 0.f, 0.f, 0.f};
    if (pr >= 0) v = *(const v4f*)(x + (size_t)(r0 - 2 + wr) * 256 + ci);
    v4bf o;
#pragma unroll
    for (int e = 0; e < 4; ++e) o[e] = (__bf16)v[e];
    *(v4bf*)(sXW + wr * 264 + ci) = o;
  }
  __syncthreads();

  // ---- stage 2: causal conv as GEMM (K=768) + bias + residual + leakyrelu ----
  {
    v8f acc[8];
    ZERO8x(acc, 8)
#pragma unroll 2
    for (int k0 = 0; k0 < 768; k0 += 32) {
      int tau = k0 >> 8;
      int i0 = k0 & 255;
      v16bf a = load_a16(sXW + (strip * 16 + m + tau) * 264 + i0, kbA);
#pragma unroll
      for (int nt = 0; nt < 8; ++nt) {
        int n = grp * 128 + nt * 16 + m;
        v16bf b = *(const v16bf*)(wcat + (size_t)n * 768 + k0 + kbB);
        acc[nt] = wmma_bf16(a, b, acc[nt]);
      }
    }
#pragma unroll
    for (int nt = 0; nt < 8; ++nt) {
      int col = grp * 128 + nt * 16 + m;
      float cb = convb[col];
#pragma unroll
      for (int j = 0; j < 8; ++j) {
        int row = strip * 16 + h16 * 8 + j;
        float yv = acc[nt][j] + cb + (float)sXW[(row + 2) * 264 + col];
        sR0[row * 260 + col] = (yv < 0.f) ? 0.01f * yv : yv;
      }
    }
  }
  __syncthreads();

  // ---- stage 3: layernorm rows of y -> sXN (bf16), vectorized ----
  {
    int row = tid >> 2, part = tid & 3;
    const float* pr = sR0 + row * 260 + part * 64;
    float s = 0.f, ss = 0.f;
#pragma unroll
    for (int cq = 0; cq < 16; ++cq) {
      v4f v = *(const v4f*)(pr + cq * 4);
#pragma unroll
      for (int e = 0; e < 4; ++e) { s += v[e]; ss += v[e] * v[e]; }
    }
    s += __shfl_xor(s, 1, 32);  ss += __shfl_xor(ss, 1, 32);
    s += __shfl_xor(s, 2, 32);  ss += __shfl_xor(ss, 2, 32);
    float mu = s * (1.f / 256.f);
    float var = ss * (1.f / 256.f) - mu * mu;
    float rs = rsqrtf(var + 1e-5f);
    __bf16* pw = sXN + row * 264 + part * 64;
#pragma unroll
    for (int cq = 0; cq < 16; ++cq) {
      v4f v = *(const v4f*)(pr + cq * 4);
      v4bf o;
#pragma unroll
      for (int e = 0; e < 4; ++e) o[e] = (__bf16)((v[e] - mu) * rs);
      *(v4bf*)(pw + cq * 4) = o;
    }
  }
  __syncthreads();

  // ---- stage 4: per-head attention ----
  for (int h = 0; h < 8; ++h) {
    // 4a: q_h = XN @ wq_h^T  (64x64); wave: strip rows, 2 col tiles
    {
      v8f acc[2];
      ZERO8x(acc, 2)
#pragma unroll 2
      for (int k0 = 0; k0 < 256; k0 += 32) {
        v16bf a = load_a16(sXN + (strip * 16 + m) * 264 + k0, kbA);
#pragma unroll
        for (int nt = 0; nt < 2; ++nt) {
          int n = h * 64 + grp * 32 + nt * 16 + m;
          v16bf b = *(const v16bf*)(wqb + (size_t)n * 256 + k0 + kbB);
          acc[nt] = wmma_bf16(a, b, acc[nt]);
        }
      }
#pragma unroll
      for (int nt = 0; nt < 2; ++nt) {
        int col = grp * 32 + nt * 16 + m;
#pragma unroll
        for (int j = 0; j < 8; ++j)
          sQH[(strip * 16 + h16 * 8 + j) * 72 + col] = (__bf16)acc[nt][j];
      }
    }
    __syncthreads();

    // 4b: scores = 0.125 * q_h @ k_h^T; wave: strip rows x 256-col half
#pragma unroll 1
    for (int nc = 0; nc < 4; ++nc) {
      v8f acc[4];
      ZERO8x(acc, 4)
#pragma unroll
      for (int k0 = 0; k0 < 64; k0 += 32) {
        v16bf a = load_a16(sQH + (strip * 16 + m) * 72 + k0, kbA);
#pragma unroll
        for (int nt = 0; nt < 4; ++nt) {
          int n = grp * 256 + nc * 64 + nt * 16 + m;
          v16bf b = *(const v16bf*)(kbuf + ((size_t)h * 512 + n) * 64 + k0 + kbB);
          acc[nt] = wmma_bf16(a, b, acc[nt]);
        }
      }
#pragma unroll
      for (int nt = 0; nt < 4; ++nt) {
        int col = grp * 256 + nc * 64 + nt * 16 + m;
#pragma unroll
        for (int j = 0; j < 8; ++j)
          sS[(strip * 16 + h16 * 8 + j) * 520 + col] =
              (__bf16)(0.125f * acc[nt][j]);
      }
    }
    __syncthreads();

    // 4c: softmax over 512 keys (4 lanes per row), vectorized v8bf;
    //     stores UNNORMALIZED exp; 1/sum folded into the P@V epilogue.
    {
      int row = tid >> 2, part = tid & 3;
      __bf16* pr = sS + row * 520 + part * 128;
      float mx = -1e30f;
#pragma unroll
      for (int c8 = 0; c8 < 16; ++c8) {
        v8bf v = *(const v8bf*)(pr + c8 * 8);
#pragma unroll
        for (int e = 0; e < 8; ++e) mx = fmaxf(mx, (float)v[e]);
      }
      mx = fmaxf(mx, __shfl_xor(mx, 1, 32));
      mx = fmaxf(mx, __shfl_xor(mx, 2, 32));
      float sum = 0.f;
#pragma unroll
      for (int c8 = 0; c8 < 16; ++c8) {
        v8bf v = *(const v8bf*)(pr + c8 * 8);
        v8bf o;
#pragma unroll
        for (int e = 0; e < 8; ++e) {
          float ex = __expf((float)v[e] - mx);
          sum += ex;
          o[e] = (__bf16)ex;
        }
        *(v8bf*)(pr + c8 * 8) = o;
      }
      sum += __shfl_xor(sum, 1, 32);
      sum += __shfl_xor(sum, 2, 32);
      if (part == 0) sInv[row] = 1.f / sum;
    }
    __syncthreads();

    // 4d: O_h = inv * (expP @ v_h) (64x64), K=512 split across grp, LDS-reduce
    {
      v8f acc[4];
      ZERO8x(acc, 4)
      const int kb0 = grp * 256;
#pragma unroll 2
      for (int k0 = 0; k0 < 256; k0 += 32) {
        v16bf a = load_a16(sS + (strip * 16 + m) * 520 + kb0 + k0, kbA);
#pragma unroll
        for (int nt = 0; nt < 4; ++nt) {
          int n = nt * 16 + m;
          v16bf b =
              *(const v16bf*)(vbuf + ((size_t)h * 64 + n) * 512 + kb0 + k0 + kbB);
          acc[nt] = wmma_bf16(a, b, acc[nt]);
        }
      }
      if (grp == 1) {
#pragma unroll
        for (int nt = 0; nt < 4; ++nt)
#pragma unroll
          for (int j = 0; j < 8; ++j)
            sRED[(strip * 16 + h16 * 8 + j) * 68 + nt * 16 + m] = acc[nt][j];
      }
      __syncthreads();
      if (grp == 0) {
#pragma unroll
        for (int nt = 0; nt < 4; ++nt) {
          int col = nt * 16 + m;
#pragma unroll
          for (int j = 0; j < 8; ++j) {
            int row = strip * 16 + h16 * 8 + j;
            sATT[row * 520 + h * 64 + col] =
                (__bf16)((acc[nt][j] + sRED[row * 68 + col]) * sInv[row]);
          }
        }
      }
      __syncthreads();
    }
  }

  // ---- stage 5: OUT = ATT @ wout^T + bout (64x256), then layernorm ----
  {
    v8f acc[8];
    ZERO8x(acc, 8)
#pragma unroll 2
    for (int k0 = 0; k0 < 512; k0 += 32) {
      v16bf a = load_a16(sATT + (strip * 16 + m) * 520 + k0, kbA);
#pragma unroll
      for (int nt = 0; nt < 8; ++nt) {
        int n = grp * 128 + nt * 16 + m;
        v16bf b = *(const v16bf*)(woutb + (size_t)n * 512 + k0 + kbB);
        acc[nt] = wmma_bf16(a, b, acc[nt]);
      }
    }
#pragma unroll
    for (int nt = 0; nt < 8; ++nt) {
      int col = grp * 128 + nt * 16 + m;
      float bo = boutp[col];
#pragma unroll
      for (int j = 0; j < 8; ++j)
        sR0[(strip * 16 + h16 * 8 + j) * 260 + col] = acc[nt][j] + bo;
    }
  }
  __syncthreads();

  // ---- final layernorm -> global, vectorized ----
  {
    int row = tid >> 2, part = tid & 3;
    const float* pr = sR0 + row * 260 + part * 64;
    float s = 0.f, ss = 0.f;
#pragma unroll
    for (int cq = 0; cq < 16; ++cq) {
      v4f v = *(const v4f*)(pr + cq * 4);
#pragma unroll
      for (int e = 0; e < 4; ++e) { s += v[e]; ss += v[e] * v[e]; }
    }
    s += __shfl_xor(s, 1, 32);  ss += __shfl_xor(ss, 1, 32);
    s += __shfl_xor(s, 2, 32);  ss += __shfl_xor(ss, 2, 32);
    float mu = s * (1.f / 256.f);
    float var = ss * (1.f / 256.f) - mu * mu;
    float rs = rsqrtf(var + 1e-5f);
    float* po = outp + (size_t)(r0 + row) * 256 + part * 64;
#pragma unroll
    for (int cq = 0; cq < 16; ++cq) {
      v4f v = *(const v4f*)(pr + cq * 4);
      v4f o;
#pragma unroll
      for (int e = 0; e < 4; ++e) o[e] = (v[e] - mu) * rs;
      *(v4f*)(po + cq * 4) = o;
    }
  }
}

// ---------------------------------------------------------------------------
extern "C" void kernel_launch(void* const* d_in, const int* in_sizes, int n_in,
                              void* d_out, int out_size, void* d_ws,
                              size_t ws_size, hipStream_t stream) {
  (void)in_sizes; (void)n_in; (void)out_size;
  const float* x     = (const float*)d_in[0];
  const float* convw = (const float*)d_in[1];
  const float* convb = (const float*)d_in[2];
  const float* pat   = (const float*)d_in[3];
  const float* wq    = (const float*)d_in[4];
  const float* wkv   = (const float*)d_in[5];
  const float* wout  = (const float*)d_in[6];
  const float* bout  = (const float*)d_in[7];
  float* out = (float*)d_out;

  char* ws = (char*)d_ws;
  float*  patln = (float*)(ws + 0);          // 512*256*4   = 524288
  __bf16* kbuf  = (__bf16*)(ws + 524288);    // 8*512*64*2  = 524288
  __bf16* vbuf  = (__bf16*)(ws + 1048576);   // 8*64*512*2  = 524288
  __bf16* wcat  = (__bf16*)(ws + 1572864);   // 256*768*2   = 393216
  __bf16* wqb   = (__bf16*)(ws + 1966080);   // 512*256*2   = 262144
  __bf16* woutb = (__bf16*)(ws + 2228224);   // 256*512*2   = 262144
  if (ws_size < 2490368) return;             // total ~2.4 MB

  sprm_prep_pat<<<2, 256, 0, stream>>>(pat, patln);
  sprm_prep_kv<<<2048, 256, 0, stream>>>(patln, wkv, kbuf, vbuf);
  sprm_prep_wt<<<1792, 256, 0, stream>>>(convw, wq, wout, wcat, wqb, woutb);
  sprm_fused<<<65536 / 64, 256, 0, stream>>>(x, convb, bout, wcat, wqb, kbuf,
                                             vbuf, woutb, out);
}
